// EdgeFeat_62027917688906
// MI455X (gfx1250) — matrix operand order repaired and checked by reading, not compiled, not run on hardware
//
#include <hip/hip_runtime.h>

typedef __attribute__((ext_vector_type(16))) _Float16 v16h;
typedef __attribute__((ext_vector_type(8)))  _Float16 v8h;
typedef __attribute__((ext_vector_type(8)))  float    v8f;

#define N_NODES   100000
#define N_EDGES   800000
#define NODE_DIM  64
#define EDGE_DIM  64
#define GEO_DIM   8
#define GEO_OUT   32
#define COND_DIM  512
#define N_GRAPHS  64
#define JOIN_DIM  96

static __device__ __forceinline__ v8f wmma16(v16h a, v16h b, v8f c) {
  // D = A(16x32 f16) * B(32x16 f16) + C(16x16 f32)
  return __builtin_amdgcn_wmma_f32_16x16x32_f16(
      /*neg_a=*/false, a, /*neg_b=*/false, b,
      /*c_mod=*/(short)0, c, /*reuse_a=*/false, /*reuse_b=*/false);
}

// ---------------------------------------------------------------------------
// Kernel 1: n_proj = node_feats @ Wn + bn   [100000,64] x [64,64]
// One wave per 16-row tile, 8 waves per block.
// ---------------------------------------------------------------------------
__global__ __launch_bounds__(256) void nproj_wmma(
    const float* __restrict__ nf, const float* __restrict__ Wn,
    const float* __restrict__ bn, float* __restrict__ nproj)
{
  __shared__ __align__(16) _Float16 wnT[64][72];  // [n][k], 144B rows (16B-aligned)
  int t = threadIdx.x;
  for (int i = t; i < 64 * 64; i += 256) {
    int n = i >> 6, k = i & 63;
    wnT[n][k] = (_Float16)Wn[k * 64 + n];
  }
  __syncthreads();

  int wave = t >> 5, lane = t & 31;
  int hsel = lane >> 4, lm = lane & 15;
  int tile = blockIdx.x * 8 + wave;
  if (tile >= N_NODES / 16) return;          // wave-uniform; EXEC stays all-1s

  const float* arow = nf + (tile * 16 + lm) * NODE_DIM;

  // A fragments: lane half h holds K in {8h..8h+7} U {16+8h..16+8h+7}
  v16h a[2];
#pragma unroll
  for (int kc = 0; kc < 2; ++kc) {
    int kb = kc * 32 + 8 * hsel;
    float tmp[16];
    ((float4*)tmp)[0] = *(const float4*)(arow + kb);
    ((float4*)tmp)[1] = *(const float4*)(arow + kb + 4);
    ((float4*)tmp)[2] = *(const float4*)(arow + kb + 16);
    ((float4*)tmp)[3] = *(const float4*)(arow + kb + 20);
#pragma unroll
    for (int e = 0; e < 16; ++e) a[kc][e] = (_Float16)tmp[e];
  }

#pragma unroll
  for (int nt = 0; nt < 4; ++nt) {
    v8f c = {0.f, 0.f, 0.f, 0.f, 0.f, 0.f, 0.f, 0.f};
#pragma unroll
    for (int kc = 0; kc < 2; ++kc) {
      // B fragment: lane half h holds K in {8h..8h+7} U {16+8h..16+8h+7}, col n = nt*16+lm
      const _Float16* bp = &wnT[nt * 16 + lm][kc * 32 + 8 * hsel];
      v8h blo = *(const v8h*)(bp);
      v8h bhi = *(const v8h*)(bp + 16);
      v16h b;
#pragma unroll
      for (int e = 0; e < 8; ++e) { b[e] = blo[e]; b[8 + e] = bhi[e]; }
      c = wmma16(a[kc], b, c);
    }
    float bias = bn[nt * 16 + lm];
    float* orow = nproj + (tile * 16 + 8 * hsel) * EDGE_DIM + nt * 16 + lm;
#pragma unroll
    for (int j = 0; j < 8; ++j) orow[j * EDGE_DIM] = c[j] + bias;  // row m=8h+j
  }
}

// ---------------------------------------------------------------------------
// Kernel 2: gb = cond @ Wc + bc   [64,512] x [512,128]  (tiny)
// ---------------------------------------------------------------------------
__global__ __launch_bounds__(256) void condproj(
    const float* __restrict__ cond, const float* __restrict__ Wc,
    const float* __restrict__ bc, float* __restrict__ gbw)
{
  int idx = blockIdx.x * 256 + threadIdx.x;
  if (idx >= N_GRAPHS * 2 * EDGE_DIM) return;
  int b = idx >> 7, o = idx & 127;
  const float* cr = cond + b * COND_DIM;
  float acc = bc[o];
  for (int k = 0; k < COND_DIM; ++k) acc = fmaf(cr[k], Wc[k * 128 + o], acc);
  gbw[idx] = acc;
}

// ---------------------------------------------------------------------------
// Kernel 3: fused gather/mul + geo proj + (join @ Wx) WMMA + FiLM + ReLU
// One wave per 16-edge tile, 8 waves per block (50000 tiles / 6250 blocks).
// ---------------------------------------------------------------------------
__global__ __launch_bounds__(256) void edge_wmma(
    const float* __restrict__ nproj, const int* __restrict__ ei,
    const float* __restrict__ egeo, const float* __restrict__ Wg,
    const float* __restrict__ bg, const float* __restrict__ gbw,
    const float* __restrict__ Wx, const int* __restrict__ ebid,
    float* __restrict__ out)
{
  __shared__ __align__(16) _Float16 wxT[64][104];        // [n][k], 208B rows
  __shared__ float gbs[N_GRAPHS * 128];
  __shared__ float wgs[GEO_DIM * GEO_OUT];
  __shared__ float bgs[GEO_OUT];
  __shared__ __align__(16) _Float16 joinb[8][16][104];   // per-wave join tile

  int t = threadIdx.x;
  for (int i = t; i < 64 * JOIN_DIM; i += 256) {
    int n = i / JOIN_DIM, k = i - n * JOIN_DIM;
    wxT[n][k] = (_Float16)Wx[k * 64 + n];
  }
  for (int i = t; i < N_GRAPHS * 128; i += 256) gbs[i] = gbw[i];
  if (t < GEO_DIM * GEO_OUT) wgs[t] = Wg[t];
  if (t < GEO_OUT) bgs[t] = bg[t];
  __syncthreads();

  int wave = t >> 5, lane = t & 31;
  int hsel = lane >> 4, lm = lane & 15;
  int tile = blockIdx.x * 8 + wave;   // 50000 tiles, grid covers exactly
  int e0 = tile * 16;

  // Phase A1: n_join = nproj[src] * nproj[dst] -> joinb[:, 0:64]
  {
    int e = e0 + lm;
    int s = ei[e], d = ei[N_EDGES + e];
    const float* sr = nproj + s * EDGE_DIM + 32 * hsel;
    const float* dr = nproj + d * EDGE_DIM + 32 * hsel;
    _Float16* jr = &joinb[wave][lm][32 * hsel];
#pragma unroll
    for (int i = 0; i < 32; i += 4) {
      float4 sv = *(const float4*)(sr + i);
      float4 dv = *(const float4*)(dr + i);
      jr[i + 0] = (_Float16)(sv.x * dv.x);
      jr[i + 1] = (_Float16)(sv.y * dv.y);
      jr[i + 2] = (_Float16)(sv.z * dv.z);
      jr[i + 3] = (_Float16)(sv.w * dv.w);
    }
  }
  // Phase A2: eg = e_geo @ Wg + bg -> joinb[:, 64:96]
  {
    int r = lane >> 1, cg = (lane & 1) * 16;
    int e = e0 + r;
    float g[8];
#pragma unroll
    for (int j = 0; j < 8; ++j) g[j] = egeo[e * GEO_DIM + j];
#pragma unroll
    for (int c = 0; c < 16; ++c) {
      int col = cg + c;
      float acc = bgs[col];
#pragma unroll
      for (int j = 0; j < 8; ++j) acc = fmaf(g[j], wgs[j * GEO_OUT + col], acc);
      joinb[wave][r][64 + col] = (_Float16)acc;
    }
  }
  __syncthreads();

  // A fragments from LDS join buffer (16B-aligned 8-half segments)
  v16h a[3];
#pragma unroll
  for (int kc = 0; kc < 3; ++kc) {
    const _Float16* p0 = &joinb[wave][lm][kc * 32 + 8 * hsel];
    const _Float16* p1 = &joinb[wave][lm][kc * 32 + 16 + 8 * hsel];
    v8h lo = *(const v8h*)p0;
    v8h hi = *(const v8h*)p1;
#pragma unroll
    for (int e = 0; e < 8; ++e) { a[kc][e] = lo[e]; a[kc][8 + e] = hi[e]; }
  }

  int bid[8];
#pragma unroll
  for (int j = 0; j < 8; ++j) bid[j] = ebid[e0 + 8 * hsel + j];

#pragma unroll
  for (int nt = 0; nt < 4; ++nt) {
    v8f c = {0.f, 0.f, 0.f, 0.f, 0.f, 0.f, 0.f, 0.f};
#pragma unroll
    for (int kc = 0; kc < 3; ++kc) {
      const _Float16* bp = &wxT[nt * 16 + lm][kc * 32 + 8 * hsel];
      v8h blo = *(const v8h*)(bp);
      v8h bhi = *(const v8h*)(bp + 16);
      v16h b;
#pragma unroll
      for (int e2 = 0; e2 < 8; ++e2) { b[e2] = blo[e2]; b[8 + e2] = bhi[e2]; }
      c = wmma16(a[kc], b, c);
    }
    int n = nt * 16 + lm;
#pragma unroll
    for (int j = 0; j < 8; ++j) {
      float gmm = gbs[bid[j] * 128 + n] + 1.0f;         // gamma + 1
      float bta = gbs[bid[j] * 128 + 64 + n];           // beta
      float v = fmaf(c[j], gmm, bta);
      out[(e0 + 8 * hsel + j) * EDGE_DIM + n] = v > 0.f ? v : 0.f;
    }
  }
}

// ---------------------------------------------------------------------------
extern "C" void kernel_launch(void* const* d_in, const int* in_sizes, int n_in,
                              void* d_out, int out_size, void* d_ws, size_t ws_size,
                              hipStream_t stream) {
  const float* nf   = (const float*)d_in[0];
  const int*   ei   = (const int*)d_in[1];
  const float* egeo = (const float*)d_in[2];
  const float* cond = (const float*)d_in[3];
  const int*   ebid = (const int*)d_in[4];
  const float* Wn   = (const float*)d_in[5];
  const float* bn   = (const float*)d_in[6];
  const float* Wg   = (const float*)d_in[7];
  const float* bg   = (const float*)d_in[8];
  const float* Wc   = (const float*)d_in[9];
  const float* bc   = (const float*)d_in[10];
  const float* Wx   = (const float*)d_in[11];
  float* out   = (float*)d_out;
  float* gbw   = (float*)d_ws;                        // [64*128]
  float* nproj = gbw + N_GRAPHS * 128;                // [100000*64]

  (void)in_sizes; (void)n_in; (void)out_size; (void)ws_size;

  nproj_wmma<<<(N_NODES / 16 + 7) / 8, 256, 0, stream>>>(nf, Wn, bn, nproj);
  condproj<<<(N_GRAPHS * 128 + 255) / 256, 256, 0, stream>>>(cond, Wc, bc, gbw);
  edge_wmma<<<(N_EDGES / 16) / 8, 256, 0, stream>>>(nproj, ei, egeo, Wg, bg,
                                                    gbw, Wx, ebid, out);
}
